// DAGCondGNNEncoder_91061896609945
// MI455X (gfx1250) — matrix operand order repaired and checked by reading, not compiled, not run on hardware
//
#include <hip/hip_runtime.h>
#include <hip/hip_bf16.h>

#define H 128
#define NBINS 128

typedef __attribute__((ext_vector_type(16))) __bf16 v16bf;
typedef __attribute__((ext_vector_type(8)))  float  v8f;

union ABReg { v16bf v; unsigned short s[16]; uint4 q[2]; };

__device__ __forceinline__ unsigned short f2bf(float f) {
    union { float f; unsigned int u; } x; x.f = f;
    unsigned int u = x.u;
    u += 0x7FFFu + ((u >> 16) & 1u);   // round-to-nearest-even
    return (unsigned short)(u >> 16);
}

// pack 4 floats -> 4 bf16 -> 8B LDS store
__device__ __forceinline__ void stage4(unsigned short* dst, float4 v) {
    unsigned int d0 = (unsigned int)f2bf(v.x) | ((unsigned int)f2bf(v.y) << 16);
    unsigned int d1 = (unsigned int)f2bf(v.z) | ((unsigned int)f2bf(v.w) << 16);
    *(uint2*)dst = make_uint2(d0, d1);
}

__device__ __forceinline__ v8f wmma_bf16(v16bf a, v16bf b, v8f c) {
    return __builtin_amdgcn_wmma_f32_16x16x32_bf16(false, a, false, b, (short)0, c, false, false);
}

// 16x32 bf16 A operand from LDS tile [rows][128 K], CDNA5 16-bit A layout.
__device__ __forceinline__ v16bf load_A_lds(const unsigned short* tile, int lane, int k0) {
    int r = lane & 15, half = lane >> 4;
    ABReg a;
    a.q[0] = *(const uint4*)(tile + r * H + k0 + half * 8);
    a.q[1] = *(const uint4*)(tile + r * H + k0 + 16 + half * 8);
    return a.v;
}

// B operand pre-packed to register layout: 8 consecutive dwords per lane.
__device__ __forceinline__ v16bf load_B_pk(const unsigned int* wpk, int nt, int kt, int lane) {
    const uint4* p = (const uint4*)(wpk + (((nt * 4 + kt) * 32 + lane) * 8));
    ABReg b; b.q[0] = p[0]; b.q[1] = p[1];
    return b.v;
}

// ---------------------------------------------------------------- utilities
__global__ void k_zero(float* p, long long n) {
    long long i = (long long)blockIdx.x * blockDim.x + threadIdx.x;
    long long st = (long long)gridDim.x * blockDim.x;
    for (; i < n; i += st) p[i] = 0.f;
}

struct W6 { const float* w[6]; };

// Pack f32 weight [K=128][N=128] -> bf16 B-operand layout per (ntile,ktile).
__global__ void k_pack(W6 ws, unsigned int* wpk) {
    int wid = blockIdx.x;            // 6 mats * 8 ntiles * 4 ktiles = 192 waves
    int mat = wid / 32;
    int nt  = (wid % 32) / 4;
    int kt  = wid % 4;
    int lane = threadIdx.x;          // 32 threads/block
    int r = lane & 15, half = lane >> 4;
    const float* W = ws.w[mat];
    unsigned int out[8];
#pragma unroll
    for (int j = 0; j < 8; j++) {
        int k = kt * 32 + half * 8 + 2 * j + (j >= 4 ? 8 : 0);
        int n = nt * 16 + r;
        unsigned int lo = f2bf(W[k * H + n]);
        unsigned int hi = f2bf(W[(k + 1) * H + n]);
        out[j] = lo | (hi << 16);
    }
    unsigned int* dst = wpk + mat * 8192 + ((nt * 4 + kt) * 32 + lane) * 8;
    *(uint4*)(dst)     = make_uint4(out[0], out[1], out[2], out[3]);
    *(uint4*)(dst + 4) = make_uint4(out[4], out[5], out[6], out[7]);
}

// tvec = time_emb @ Wt + bt   (tiny, 1 block)
__global__ void k_tvec(const float* t, const float* Wt, const float* bt, float* tvec) {
    int n = threadIdx.x;
    float acc = bt[n];
    for (int k = 0; k < H; k++) acc += t[k] * Wt[k * H + n];
    tvec[n] = acc;
}

// ------------- node GEMMs (U,V,A,B): one 32-row h tile feeds 4 matrices
struct NG { const float* h; const float* bias[4]; float* out[4]; };

__global__ void k_node_gemm(NG p, const unsigned int* wpk, int V) {
    __shared__ __align__(16) unsigned short tile[32 * H];
    int m0 = blockIdx.x * 32;
    int tid = threadIdx.x;
    for (int idx4 = tid; idx4 < 32 * (H / 4); idx4 += 256) {
        int rl = idx4 >> 5;                    // 32 float4 groups per row
        int c4 = (idx4 & 31) << 2;
        int row = m0 + rl; if (row >= V) row = V - 1;
        stage4(tile + rl * H + c4, *(const float4*)(p.h + (size_t)row * H + c4));
    }
    __syncthreads();
    int wave = tid >> 5, lane = tid & 31;
    int nt = wave;                             // 8 waves -> 8 N-tiles
    int r = lane & 15, half = lane >> 4;
    int n = nt * 16 + r;
    bool full = (m0 + 32 <= V);
#pragma unroll
    for (int mat = 0; mat < 4; mat++) {
        const unsigned int* wm = wpk + mat * 8192;
        v8f acc0 = {}, acc1 = {};
#pragma unroll
        for (int kt = 0; kt < 4; kt++) {
            v16bf b  = load_B_pk(wm, nt, kt, lane);
            v16bf a0 = load_A_lds(tile, lane, kt * 32);
            v16bf a1 = load_A_lds(tile + 16 * H, lane, kt * 32);
            acc0 = wmma_bf16(a0, b, acc0);
            acc1 = wmma_bf16(a1, b, acc1);
        }
        float bb = p.bias[mat][n];
        float* out0 = p.out[mat] + (size_t)(m0 + 8 * half) * H + n;
        float* out1 = out0 + 16 * H;
        if (full) {
#pragma unroll
            for (int i = 0; i < 8; i++) out0[i * H] = acc0[i] + bb;
#pragma unroll
            for (int i = 0; i < 8; i++) out1[i * H] = acc1[i] + bb;
        } else {
#pragma unroll
            for (int i = 0; i < 8; i++) {
                int row = m0 + i + 8 * half;
                if (row < V)      out0[i * H] = acc0[i] + bb;
                if (row + 16 < V) out1[i * H] = acc1[i] + bb;
            }
        }
    }
}

// --------- edge phase 1 epilogue (templated full/partial tile)
template <bool FULL>
__device__ __forceinline__ void edge_pre_epi(
        const v8f* accs, float bb, const int* sridx, const int* scidx,
        const float* Ah, const float* Bh, const float* Vhp,
        float* epre, float* agg, int m0, int half, int n, int E,
        float& s1, float& s2) {
#pragma unroll
    for (int sub = 0; sub < 2; sub++) {
#pragma unroll
        for (int i = 0; i < 8; i++) {
            int m = sub * 16 + i + 8 * half;
            int eid = m0 + m;
            if (!FULL && eid >= E) continue;
            int ri = sridx[m], ci = scidx[m];
            float pre = accs[sub][i] + bb + Ah[(size_t)ci * H + n]
                                          + Bh[(size_t)ri * H + n];
            float gate = 1.f / (1.f + __expf(-pre));
            float gv = gate * Vhp[(size_t)ci * H + n];
            epre[(size_t)eid * H + n] = pre;
            unsafeAtomicAdd(&agg[(size_t)ri * H + n], gv);   // L2-resident
            s1 += pre; s2 += pre * pre;
        }
    }
}

// --------- edge phase 1: Ce GEMM + gathers + gates + scatter-add + e-stats
__global__ void k_edge_pre(const float* e, const int* eidx, const float* bC,
                           const float* Ah, const float* Bh, const float* Vhp,
                           const unsigned int* wpkC, float* epre, float* agg,
                           float* ebins, int E) {
    __shared__ __align__(16) unsigned short tile[32 * H];
    __shared__ int sridx[32], scidx[32];
    __shared__ float lsum[H], lsq[H];
    int m0 = blockIdx.x * 32;
    int tid = threadIdx.x;
    if (tid < 32) {
        int eid = m0 + tid; if (eid >= E) eid = E - 1;
        sridx[tid] = eidx[eid];                 // edge_index[0] : scatter row
        scidx[tid] = eidx[(size_t)E + eid];     // edge_index[1] : gather col
    }
    if (tid < H) { lsum[tid] = 0.f; lsq[tid] = 0.f; }
    for (int idx4 = tid; idx4 < 32 * (H / 4); idx4 += 256) {
        int rl = idx4 >> 5;
        int c4 = (idx4 & 31) << 2;
        int row = m0 + rl; if (row >= E) row = E - 1;
        stage4(tile + rl * H + c4, *(const float4*)(e + (size_t)row * H + c4));
    }
    if (tid == 0 && m0 + 32 < E) __builtin_prefetch(e + (size_t)(m0 + 32) * H, 0, 1);
    __syncthreads();
    int wave = tid >> 5, lane = tid & 31;
    int nt = wave;
    v8f accs[2]; accs[0] = {}; accs[1] = {};
#pragma unroll
    for (int kt = 0; kt < 4; kt++) {
        v16bf b  = load_B_pk(wpkC, nt, kt, lane);
        v16bf a0 = load_A_lds(tile, lane, kt * 32);
        v16bf a1 = load_A_lds(tile + 16 * H, lane, kt * 32);
        accs[0] = wmma_bf16(a0, b, accs[0]);
        accs[1] = wmma_bf16(a1, b, accs[1]);
    }
    int r = lane & 15, half = lane >> 4;
    int n = nt * 16 + r;                        // fixed channel per lane
    float bb = bC[n];
    float s1 = 0.f, s2 = 0.f;
    if (m0 + 32 <= E)
        edge_pre_epi<true >(accs, bb, sridx, scidx, Ah, Bh, Vhp, epre, agg, m0, half, n, E, s1, s2);
    else
        edge_pre_epi<false>(accs, bb, sridx, scidx, Ah, Bh, Vhp, epre, agg, m0, half, n, E, s1, s2);
    atomicAdd(&lsum[n], s1);                    // ds_add_f32
    atomicAdd(&lsq[n], s2);
    __syncthreads();
    if (tid < H) {
        int bin = blockIdx.x & (NBINS - 1);
        unsafeAtomicAdd(&ebins[(size_t)bin * H + tid], lsum[tid]);
        unsafeAtomicAdd(&ebins[(size_t)NBINS * H + (size_t)bin * H + tid], lsq[tid]);
    }
}

// ------------------------------------------- node BN stats of (Uh + agg)
__global__ void k_node_stats(const float* Uh, const float* agg, float* hbins, int V) {
    int tid = threadIdx.x;
    int n = tid & 127, rg = tid >> 7;
    float s1 = 0.f, s2 = 0.f;
    int base = blockIdx.x * 64;
    for (int rr = rg; rr < 64; rr += 2) {
        int row = base + rr;
        if (row < V) {
            float y = Uh[(size_t)row * H + n] + agg[(size_t)row * H + n];
            s1 += y; s2 += y * y;
        }
    }
    int bin = blockIdx.x & (NBINS - 1);
    unsafeAtomicAdd(&hbins[(size_t)bin * H + n], s1);
    unsafeAtomicAdd(&hbins[(size_t)NBINS * H + (size_t)bin * H + n], s2);
}

// mean/var -> fused scale/shift (stat: sc_e, sh_e, sc_h, sh_h)
__global__ void k_finalize(const float* ebins, const float* hbins,
                           const float* g_e, const float* b_e,
                           const float* g_h, const float* b_h,
                           float* stat, float fE, float fV) {
    int n = threadIdx.x;    // 128
    float se1 = 0, se2 = 0, sh1 = 0, sh2 = 0;
    for (int b = 0; b < NBINS; b++) {
        se1 += ebins[b * H + n]; se2 += ebins[NBINS * H + b * H + n];
        sh1 += hbins[b * H + n]; sh2 += hbins[NBINS * H + b * H + n];
    }
    float me = se1 / fE, ve = se2 / fE - me * me;
    float sce = g_e[n] * rsqrtf(ve + 1e-5f);
    stat[n] = sce;
    stat[H + n] = b_e[n] - me * sce;
    float mh = sh1 / fV, vh = sh2 / fV - mh * mh;
    float sch = g_h[n] * rsqrtf(vh + 1e-5f);
    stat[2 * H + n] = sch;
    stat[3 * H + n] = b_h[n] - mh * sch;
}

// x_out = h + relu(bn(Uh + agg))    (float4 streaming)
__global__ void k_node_out(const float* h, const float* Uh, const float* agg,
                           const float* stat, float* xout, long long n4) {
    long long i = (long long)blockIdx.x * blockDim.x + threadIdx.x;
    long long st = (long long)gridDim.x * blockDim.x;
    for (; i < n4; i += st) {
        long long e0 = i * 4;
        int c = (int)(e0 & 127);
        float4 u  = *(const float4*)(Uh + e0);
        float4 a  = *(const float4*)(agg + e0);
        float4 hh = *(const float4*)(h + e0);
        float4 sc = *(const float4*)(stat + 2 * H + c);
        float4 sh = *(const float4*)(stat + 3 * H + c);
        float4 o;
        o.x = hh.x + fmaxf((u.x + a.x) * sc.x + sh.x, 0.f);
        o.y = hh.y + fmaxf((u.y + a.y) * sc.y + sh.y, 0.f);
        o.z = hh.z + fmaxf((u.z + a.z) * sc.z + sh.z, 0.f);
        o.w = hh.w + fmaxf((u.w + a.w) * sc.w + sh.w, 0.f);
        *(float4*)(xout + e0) = o;
    }
}

template <bool FULL>
__device__ __forceinline__ void edge_out_epi(
        const v8f* accs, float bb, const float* e, float* ebuf,
        int m0, int half, int n, int E) {
#pragma unroll
    for (int sub = 0; sub < 2; sub++) {
        const float* src = e    + (size_t)(m0 + sub * 16 + 8 * half) * H + n;
        float*       dst = ebuf + (size_t)(m0 + sub * 16 + 8 * half) * H + n;
#pragma unroll
        for (int i = 0; i < 8; i++) {
            if (!FULL && m0 + sub * 16 + 8 * half + i >= E) continue;
            dst[i * H] = src[i * H] + accs[sub][i] + bb;
        }
    }
}

// e_out = e + (relu(bn(e_pre)) + tvec) @ Wo + bo ; e_pre lives in the output
// buffer and is overwritten in place (reads precede writes via block barrier).
__global__ void k_edge_out(const float* e, const float* bo, const unsigned int* wpkO,
                           const float* stat, const float* tvec,
                           float* ebuf, int E) {
    __shared__ __align__(16) unsigned short tile[32 * H];
    __shared__ float ssc[H], ssh[H], stv[H];
    int m0 = blockIdx.x * 32;
    int tid = threadIdx.x;
    if (tid < H) { ssc[tid] = stat[tid]; ssh[tid] = stat[H + tid]; stv[tid] = tvec[tid]; }
    __syncthreads();
    for (int idx4 = tid; idx4 < 32 * (H / 4); idx4 += 256) {
        int rl = idx4 >> 5;
        int c4 = (idx4 & 31) << 2;
        int row = m0 + rl; if (row >= E) row = E - 1;
        float4 p  = *(const float4*)(ebuf + (size_t)row * H + c4);
        float4 sc = *(const float4*)(ssc + c4);
        float4 sh = *(const float4*)(ssh + c4);
        float4 tv = *(const float4*)(stv + c4);
        float4 v;
        v.x = fmaxf(p.x * sc.x + sh.x, 0.f) + tv.x;
        v.y = fmaxf(p.y * sc.y + sh.y, 0.f) + tv.y;
        v.z = fmaxf(p.z * sc.z + sh.z, 0.f) + tv.z;
        v.w = fmaxf(p.w * sc.w + sh.w, 0.f) + tv.w;
        stage4(tile + rl * H + c4, v);
    }
    __syncthreads();    // all e_pre reads of this block's rows are done
    int wave = tid >> 5, lane = tid & 31, nt = wave;
    v8f accs[2]; accs[0] = {}; accs[1] = {};
#pragma unroll
    for (int kt = 0; kt < 4; kt++) {
        v16bf b  = load_B_pk(wpkO, nt, kt, lane);
        v16bf a0 = load_A_lds(tile, lane, kt * 32);
        v16bf a1 = load_A_lds(tile + 16 * H, lane, kt * 32);
        accs[0] = wmma_bf16(a0, b, accs[0]);
        accs[1] = wmma_bf16(a1, b, accs[1]);
    }
    int r = lane & 15, half = lane >> 4, n = nt * 16 + r;
    float bb = bo[n];
    if (m0 + 32 <= E)
        edge_out_epi<true >(accs, bb, e, ebuf, m0, half, n, E);
    else
        edge_out_epi<false>(accs, bb, e, ebuf, m0, half, n, E);
}

extern "C" void kernel_launch(void* const* d_in, const int* in_sizes, int n_in,
                              void* d_out, int out_size, void* d_ws, size_t ws_size,
                              hipStream_t stream) {
    const float* h   = (const float*)d_in[0];
    const float* e   = (const float*)d_in[1];
    const float* te  = (const float*)d_in[2];
    const int*   ei  = (const int*)d_in[3];
    const float* WU  = (const float*)d_in[4];  const float* bU = (const float*)d_in[5];
    const float* WV  = (const float*)d_in[6];  const float* bV = (const float*)d_in[7];
    const float* WA  = (const float*)d_in[8];  const float* bA = (const float*)d_in[9];
    const float* WB  = (const float*)d_in[10]; const float* bB = (const float*)d_in[11];
    const float* WC  = (const float*)d_in[12]; const float* bC = (const float*)d_in[13];
    const float* Wt  = (const float*)d_in[14]; const float* bt = (const float*)d_in[15];
    const float* Wo  = (const float*)d_in[16]; const float* bo = (const float*)d_in[17];
    const float* g_h = (const float*)d_in[18]; const float* b_h = (const float*)d_in[19];
    const float* g_e = (const float*)d_in[20]; const float* b_e = (const float*)d_in[21];

    int V = in_sizes[0] / H;
    int E = in_sizes[1] / H;
    size_t VH = (size_t)V * H;

    float* ws    = (float*)d_ws;
    float* Uh    = ws;
    float* Vhp   = ws + VH;
    float* Ah    = ws + 2 * VH;
    float* Bh    = ws + 3 * VH;
    float* agg   = ws + 4 * VH;
    float* ebins = ws + 5 * VH;                      // 2*NBINS*H
    float* hbins = ebins + 2 * (size_t)NBINS * H;    // 2*NBINS*H
    float* stat  = hbins + 2 * (size_t)NBINS * H;    // 4*H
    float* tvec  = stat + 4 * H;                     // H
    unsigned int* wpk = (unsigned int*)(tvec + H);   // 6 * 8192 dwords

    float* xout    = (float*)d_out;
    float* eoutbuf = (float*)d_out + VH;             // e_pre then e_out

    long long zn = (long long)VH + 4LL * NBINS * H;  // agg + both bin arrays
    k_zero<<<1024, 256, 0, stream>>>(agg, zn);

    W6 w6; w6.w[0] = WU; w6.w[1] = WV; w6.w[2] = WA; w6.w[3] = WB; w6.w[4] = WC; w6.w[5] = Wo;
    k_pack<<<192, 32, 0, stream>>>(w6, wpk);
    k_tvec<<<1, H, 0, stream>>>(te, Wt, bt, tvec);

    NG ng; ng.h = h;
    ng.bias[0] = bU; ng.bias[1] = bV; ng.bias[2] = bA; ng.bias[3] = bB;
    ng.out[0] = Uh;  ng.out[1] = Vhp; ng.out[2] = Ah; ng.out[3] = Bh;
    k_node_gemm<<<(V + 31) / 32, 256, 0, stream>>>(ng, wpk, V);

    int mE = (E + 31) / 32;
    k_edge_pre<<<mE, 256, 0, stream>>>(e, ei, bC, Ah, Bh, Vhp,
                                       wpk + 4 * 8192, eoutbuf, agg, ebins, E);

    k_node_stats<<<(V + 63) / 64, 256, 0, stream>>>(Uh, agg, hbins, V);
    k_finalize<<<1, H, 0, stream>>>(ebins, hbins, g_e, b_e, g_h, b_h,
                                    stat, (float)E, (float)V);
    k_node_out<<<2048, 256, 0, stream>>>(h, Uh, agg, stat, xout, (long long)(VH / 4));
    k_edge_out<<<mE, 256, 0, stream>>>(e, bo, wpk + 5 * 8192, stat, tvec, eoutbuf, E);
}
